// GRUCuda_80187039416533
// MI455X (gfx1250) — matrix operand order repaired and checked by reading, not compiled, not run on hardware
//
#include <hip/hip_runtime.h>
#include <hip/hip_bf16.h>

// GRU: B=64 T=512 IN=1024 H=1024 NH=4 HD=256 NG=3
// Phase 1: Wx = input @ W + bW  (f16 WMMA, f32 accum) -> ws
// Phase 2: 16 persistent WGs (4 batch-tiles x 4 heads), h in LDS,
//          per-step recurrent GEMM via v_wmma_f32_16x16x32_f16.
//          Wx loads hoisted ahead of the WMMA chain to hide memory latency
//          on the sequential critical path.

typedef __attribute__((ext_vector_type(16))) _Float16 v16h;
typedef __attribute__((ext_vector_type(8)))  _Float16 v8h;
typedef __attribute__((ext_vector_type(8)))  float    v8f;

union V16U { v16h v; v8h h[2]; };

#define BATCH 64
#define SEQ   512
#define INDIM 1024
#define HID   1024
#define NHEAD 4
#define HDIM  256
#define NCOLS 3072   // NG*H
#define MROWS (BATCH*SEQ)

__device__ static inline float fast_sigmoid(float x) {
  return 1.0f / (1.0f + __expf(-x));
}
__device__ static inline float fast_tanh(float x) {
  // tanh(x) = 2*sigmoid(2x) - 1 ; v_exp_f32 path, no libm call
  return 2.0f / (1.0f + __expf(-2.0f * x)) - 1.0f;
}

// ---- fragment loaders (ISA 7.12.2 layouts) -------------------------------

// A-matrix 16x32 f16: lane<16: row=lane,  K={kb..kb+7, kb+16..kb+23}, kb=0
//                     lane>=16: row=lane-16, same with kb=8
__device__ static inline v16h frag_a_g(const _Float16* __restrict__ tile, int ld, int lane) {
  const int row = lane & 15;
  const int kb  = (lane >> 4) << 3;        // 0 or 8
  V16U u;
  u.h[0] = *(const v8h*)(tile + (size_t)row * ld + kb);
  u.h[1] = *(const v8h*)(tile + (size_t)row * ld + kb + 16);
  return u.v;
}

// B-matrix 32x16 f16 (tile stored as [col][k], k contiguous):
// lanes 0-15: col=lane, K={0..7,16..23} ; lanes 16-31: col=lane-16, K={8..15,24..31}
__device__ static inline v16h frag_b_g(const _Float16* __restrict__ tile, int ld, int lane) {
  const int col = lane & 15;
  const int kb  = (lane >> 4) << 3;        // 0 or 8
  V16U u;
  u.h[0] = *(const v8h*)(tile + (size_t)col * ld + kb);
  u.h[1] = *(const v8h*)(tile + (size_t)col * ld + kb + 16);
  return u.v;
}

// ---- conversion kernels ---------------------------------------------------

__global__ void cvt_f16_kernel(const float* __restrict__ in, _Float16* __restrict__ out, int n) {
  for (int i = blockIdx.x * blockDim.x + threadIdx.x; i < n; i += gridDim.x * blockDim.x)
    out[i] = (_Float16)in[i];
}

// W [IN, 3072] (row-major) -> Wt [3072, IN] f16 (row-major in K)
__global__ void cvt_wt_kernel(const float* __restrict__ W, _Float16* __restrict__ Wt) {
  for (int i = blockIdx.x * blockDim.x + threadIdx.x; i < NCOLS * INDIM;
       i += gridDim.x * blockDim.x) {
    int n = i >> 10;          // 0..3071
    int k = i & 1023;
    Wt[i] = (_Float16)W[(size_t)k * NCOLS + n];
  }
}

// R [NH, HD(i), NG, HD(o)] -> Rt [NH, NG, HD(o), HD(i)] f16
__global__ void cvt_r_kernel(const float* __restrict__ R, _Float16* __restrict__ Rt) {
  for (int idx = blockIdx.x * blockDim.x + threadIdx.x; idx < NHEAD * 3 * HDIM * HDIM;
       idx += gridDim.x * blockDim.x) {
    int head = idx / (3 * HDIM * HDIM);
    int rem  = idx % (3 * HDIM * HDIM);
    int g    = rem / (HDIM * HDIM);
    int rem2 = rem % (HDIM * HDIM);
    int o    = rem2 >> 8;
    int i    = rem2 & 255;
    Rt[idx] = (_Float16)R[(((size_t)head * HDIM + i) * 3 + g) * HDIM + o];
  }
}

// ---- Phase 1: Wx = Ain(f16) @ Wt^T(f16) + bW  ----------------------------
// Block: 256 thr = 8 waves (4 M x 2 N). WG tile 128x64, wave tile 32x32.
__global__ __launch_bounds__(256)
void gemm_wx_kernel(const _Float16* __restrict__ Ain, const _Float16* __restrict__ Wt,
                    const float* __restrict__ bW, float* __restrict__ Wx) {
  const int lane = threadIdx.x & 31;
  const int w    = threadIdx.x >> 5;
  const int m0   = blockIdx.y * 128 + (w & 3) * 32;
  const int n0   = blockIdx.x * 64  + (w >> 2) * 32;

  v8f acc00 = {}, acc01 = {}, acc10 = {}, acc11 = {};
  const _Float16* a0p = Ain + (size_t)m0 * INDIM;
  const _Float16* a1p = Ain + (size_t)(m0 + 16) * INDIM;
  const _Float16* b0p = Wt  + (size_t)n0 * INDIM;
  const _Float16* b1p = Wt  + (size_t)(n0 + 16) * INDIM;

  for (int k0 = 0; k0 < INDIM; k0 += 32) {
    v16h a0 = frag_a_g(a0p + k0, INDIM, lane);
    v16h a1 = frag_a_g(a1p + k0, INDIM, lane);
    v16h b0 = frag_b_g(b0p + k0, INDIM, lane);
    v16h b1 = frag_b_g(b1p + k0, INDIM, lane);
    acc00 = __builtin_amdgcn_wmma_f32_16x16x32_f16(false, a0, false, b0, (short)0, acc00, false, false);
    acc01 = __builtin_amdgcn_wmma_f32_16x16x32_f16(false, a0, false, b1, (short)0, acc01, false, false);
    acc10 = __builtin_amdgcn_wmma_f32_16x16x32_f16(false, a1, false, b0, (short)0, acc10, false, false);
    acc11 = __builtin_amdgcn_wmma_f32_16x16x32_f16(false, a1, false, b1, (short)0, acc11, false, false);
  }

  // C layout: VGPR r, lane L -> M = r + (L>=16 ? 8 : 0), N = L&15
  const int mo = (lane >> 4) << 3;
  const int nc = lane & 15;
  const float bias0 = bW[n0 + nc];
  const float bias1 = bW[n0 + 16 + nc];
#pragma unroll
  for (int r = 0; r < 8; ++r) {
    const size_t m  = (size_t)(m0 + mo + r);
    Wx[m * NCOLS + n0 + nc]             = acc00[r] + bias0;
    Wx[m * NCOLS + n0 + 16 + nc]        = acc01[r] + bias1;
    Wx[(m + 16) * NCOLS + n0 + nc]      = acc10[r] + bias0;
    Wx[(m + 16) * NCOLS + n0 + 16 + nc] = acc11[r] + bias1;
  }
}

// ---- Phase 2: sequential GRU scan ----------------------------------------
// 16 WGs: blockIdx.x = bt*4 + head. 256 thr = 8 waves; wave w owns o-chunks
// {2w, 2w+1} (16 outputs each) for all 3 gates: 6 WMMA column tiles.
#define HSTRIDE 272   // padded f16 h row stride (16B-aligned, bank-skewed)

__global__ __launch_bounds__(256)
void gru_scan_kernel(const float* __restrict__ Wx, const _Float16* __restrict__ Rt,
                     const float* __restrict__ bias, float* __restrict__ out) {
  __shared__ float    hF32[16 * HDIM];
  __shared__ _Float16 hF16[16 * HSTRIDE];

  const int lane = threadIdx.x & 31;
  const int w    = threadIdx.x >> 5;
  const int head = blockIdx.x & 3;
  const int bt   = blockIdx.x >> 2;

  for (int i = threadIdx.x; i < 16 * HDIM;    i += blockDim.x) hF32[i] = 0.0f;
  for (int i = threadIdx.x; i < 16 * HSTRIDE; i += blockDim.x) hF16[i] = (_Float16)0.0f;
  __syncthreads();

  const int c0 = w * 2;
  const int nc = lane & 15;
  const int mo = (lane >> 4) << 3;

  float bz[2], br[2], bn[2];
#pragma unroll
  for (int c = 0; c < 2; ++c) {
    const int o = (c0 + c) * 16 + nc;
    bz[c] = bias[(head * 3 + 0) * HDIM + o];
    br[c] = bias[(head * 3 + 1) * HDIM + o];
    bn[c] = bias[(head * 3 + 2) * HDIM + o];
  }

  const _Float16* __restrict__ Rh = Rt + (size_t)head * 3 * HDIM * HDIM;

  for (int t = 0; t < SEQ; ++t) {
    // ---- hoisted Wx preactivation loads (independent of h): issue first so
    // their latency overlaps the LDS fragment reads and the WMMA chain.
    float xz[2][8], xr[2][8], xn[2][8];
#pragma unroll
    for (int c = 0; c < 2; ++c) {
#pragma unroll
      for (int r = 0; r < 8; ++r) {
        const int bglob = bt * 16 + mo + r;
        const int o = (c0 + c) * 16 + nc;
        const float* __restrict__ p =
            Wx + ((size_t)bglob * SEQ + t) * NCOLS + head * HDIM + o;
        xz[c][r] = p[0];
        xr[c][r] = p[1024];
        xn[c][r] = p[2048];
      }
    }

    // ---- A fragments of h from LDS (8 K-tiles of 32)
    v16h av[8];
    {
      const int row = lane & 15;
      const int kb  = (lane >> 4) << 3;
#pragma unroll
      for (int kt = 0; kt < 8; ++kt) {
        V16U u;
        u.h[0] = *(const v8h*)(&hF16[row * HSTRIDE + kt * 32 + kb]);
        u.h[1] = *(const v8h*)(&hF16[row * HSTRIDE + kt * 32 + kb + 16]);
        av[kt] = u.v;
      }
    }

    v8f az[2] = {{}, {}}, ar[2] = {{}, {}}, an[2] = {{}, {}};
#pragma unroll
    for (int kt = 0; kt < 8; ++kt) {
      const int k0 = kt * 32;
#pragma unroll
      for (int c = 0; c < 2; ++c) {
        const int ocol = (c0 + c) * 16;
        v16h fz = frag_b_g(Rh + ((size_t)(0 * HDIM + ocol)) * HDIM + k0, HDIM, lane);
        v16h fr = frag_b_g(Rh + ((size_t)(1 * HDIM + ocol)) * HDIM + k0, HDIM, lane);
        v16h fn = frag_b_g(Rh + ((size_t)(2 * HDIM + ocol)) * HDIM + k0, HDIM, lane);
        az[c] = __builtin_amdgcn_wmma_f32_16x16x32_f16(false, av[kt], false, fz, (short)0, az[c], false, false);
        ar[c] = __builtin_amdgcn_wmma_f32_16x16x32_f16(false, av[kt], false, fr, (short)0, ar[c], false, false);
        an[c] = __builtin_amdgcn_wmma_f32_16x16x32_f16(false, av[kt], false, fn, (short)0, an[c], false, false);
      }
    }

    __syncthreads();  // everyone done reading hF16 before we overwrite h

#pragma unroll
    for (int c = 0; c < 2; ++c) {
#pragma unroll
      for (int r = 0; r < 8; ++r) {
        const int ml = mo + r;                 // local batch row 0..15
        const int bglob = bt * 16 + ml;
        const int o  = (c0 + c) * 16 + nc;
        const float zf = fast_sigmoid(xz[c][r] + az[c][r] + bz[c]);
        const float rf = fast_sigmoid(xr[c][r] + ar[c][r] + br[c]);
        const float nf = fast_tanh(xn[c][r] + rf * an[c][r] + bn[c]);
        const float hold = hF32[ml * HDIM + o];
        const float hnew = zf * hold + (1.0f - zf) * nf;
        hF32[ml * HDIM + o]    = hnew;
        hF16[ml * HSTRIDE + o] = (_Float16)hnew;
        out[((size_t)bglob * SEQ + t) * HID + head * HDIM + o] = hnew;
      }
    }
    __syncthreads();  // h update visible before next step's fragment loads
  }

  // ---- final hidden state hn = h[T-1]  (read back from LDS master copy)
  float* __restrict__ outHn = out + (size_t)BATCH * SEQ * HID;
#pragma unroll
  for (int c = 0; c < 2; ++c) {
#pragma unroll
    for (int r = 0; r < 8; ++r) {
      const int ml = mo + r;
      const int o  = (c0 + c) * 16 + nc;
      outHn[(size_t)(bt * 16 + ml) * HID + head * HDIM + o] = hF32[ml * HDIM + o];
    }
  }
}

// ---- launcher -------------------------------------------------------------

extern "C" void kernel_launch(void* const* d_in, const int* in_sizes, int n_in,
                              void* d_out, int out_size, void* d_ws, size_t ws_size,
                              hipStream_t stream) {
  const float* inp = (const float*)d_in[0];   // [B, T, IN]
  const float* W   = (const float*)d_in[1];   // [IN, 3H]
  const float* bW  = (const float*)d_in[2];   // [3H]
  const float* R   = (const float*)d_in[3];   // [NH, HD, NG, HD]
  const float* b   = (const float*)d_in[4];   // [NH, NG, HD]
  float* out = (float*)d_out;

  char* ws = (char*)d_ws;
  const size_t wx_bytes  = (size_t)MROWS * NCOLS * sizeof(float);        // 402,653,184
  const size_t ain_bytes = (size_t)MROWS * INDIM * sizeof(_Float16);     //  67,108,864
  const size_t wt_bytes  = (size_t)NCOLS * INDIM * sizeof(_Float16);     //   6,291,456
  float*    Wx  = (float*)ws;
  _Float16* Ain = (_Float16*)(ws + wx_bytes);
  _Float16* Wt  = (_Float16*)(ws + wx_bytes + ain_bytes);
  _Float16* Rt  = (_Float16*)(ws + wx_bytes + ain_bytes + wt_bytes);

  cvt_f16_kernel<<<2048, 256, 0, stream>>>(inp, Ain, MROWS * INDIM);
  cvt_wt_kernel<<<2048, 256, 0, stream>>>(W, Wt);
  cvt_r_kernel<<<768, 256, 0, stream>>>(R, Rt);

  dim3 g1(NCOLS / 64, MROWS / 128);           // 48 x 256
  gemm_wx_kernel<<<g1, 256, 0, stream>>>(Ain, Wt, bW, Wx);

  gru_scan_kernel<<<16, 256, 0, stream>>>(Wx, Rt, b, out);
}
